// DDiTBlock_63462436765839
// MI455X (gfx1250) — compile-verified, matrix operand order checked by reading, not run on hardware
//
#include <hip/hip_runtime.h>
#include <hip/hip_bf16.h>
#include <math.h>

// ---------------- problem constants ----------------
#define D_     768
#define S_     2048
#define B_     4
#define H_     12
#define DH_    64
#define MLP_   3072
#define SIX_   (6 * D_)        // 4608
#define MROWS  (B_ * S_)       // 8192

typedef _Float16 h16;
typedef __attribute__((ext_vector_type(16))) _Float16 v16h;
typedef __attribute__((ext_vector_type(8)))  float    v8f;
typedef __attribute__((ext_vector_type(4)))  float    f4;

union Frag { v16h h; f4 f[2]; };

__device__ __forceinline__ v8f wmma_f16(v16h a, v16h b, v8f c) {
    // D = A(16x32 f16) * B(32x16 f16) + C(16x16 f32)
    return __builtin_amdgcn_wmma_f32_16x16x32_f16(false, a, false, b, (short)0, c,
                                                  false, false);
}

// ---- DPP16 cross-lane ops: butterfly reduce across 16-lane row, all in VALU ----
template <int CTRL>
__device__ __forceinline__ float dppf(float x) {
    int i = __builtin_bit_cast(int, x);
    int r = __builtin_amdgcn_update_dpp(0, i, CTRL, 0xf, 0xf, true);
    return __builtin_bit_cast(float, r);
}
__device__ __forceinline__ float red16_max(float v) {
    v = fmaxf(v, dppf<0xB1>(v));    // quad_perm(1,0,3,2)
    v = fmaxf(v, dppf<0x4E>(v));    // quad_perm(2,3,0,1)
    v = fmaxf(v, dppf<0x141>(v));   // row_half_mirror
    v = fmaxf(v, dppf<0x140>(v));   // row_mirror
    return v;
}
__device__ __forceinline__ float red16_sum(float v) {
    v += dppf<0xB1>(v);
    v += dppf<0x4E>(v);
    v += dppf<0x141>(v);
    v += dppf<0x140>(v);
    return v;
}

// ---------------- f32 -> f16 convert ----------------
__global__ void k_f32_to_f16(const float* __restrict__ in, h16* __restrict__ out, int n) {
    int i = blockIdx.x * blockDim.x + threadIdx.x;
    if (i < n) out[i] = (h16)in[i];
}

// ---------------- ada = c @ W_ada^T + b_ada ----------------
__global__ void k_ada(const float* __restrict__ c, const float* __restrict__ Wada,
                      const float* __restrict__ bada, float* __restrict__ ada) {
    int i = blockIdx.x * blockDim.x + threadIdx.x;
    if (i >= B_ * SIX_) return;
    int b = i / SIX_, j = i % SIX_;
    const float* wr = Wada + (size_t)j * 128;
    const float* cb = c + b * 128;
    float s = bada[j];
#pragma unroll 8
    for (int k = 0; k < 128; k++) s = fmaf(cb[k], wr[k], s);
    ada[i] = s;
}

// ---------------- LayerNorm + adaLN modulation, write f16 ----------------
__global__ __launch_bounds__(256) void k_ln_mod(const float* __restrict__ x,
                                                const float* __restrict__ w,
                                                const float* __restrict__ ada,
                                                int shOff, int scOff,
                                                h16* __restrict__ out) {
    __shared__ float red[256];
    const int row = blockIdx.x;
    const int b   = row >> 11;            // row / S
    const int tid = threadIdx.x;
    const float* xr = x + (size_t)row * D_;
    float v0 = xr[tid], v1 = xr[tid + 256], v2 = xr[tid + 512];

    red[tid] = v0 + v1 + v2;
    __syncthreads();
    for (int o = 128; o > 0; o >>= 1) { if (tid < o) red[tid] += red[tid + o]; __syncthreads(); }
    float mu = red[0] * (1.0f / D_);
    __syncthreads();

    float d0 = v0 - mu, d1 = v1 - mu, d2 = v2 - mu;
    red[tid] = d0 * d0 + d1 * d1 + d2 * d2;
    __syncthreads();
    for (int o = 128; o > 0; o >>= 1) { if (tid < o) red[tid] += red[tid + o]; __syncthreads(); }
    float rstd = rsqrtf(red[0] * (1.0f / D_) + 1e-5f);

    const float* adb = ada + b * SIX_;
    h16* orow = out + (size_t)row * D_;
    float dv[3] = {d0, d1, d2};
    int d = tid;
#pragma unroll
    for (int j = 0; j < 3; j++, d += 256) {
        float nv = dv[j] * rstd * w[d];
        orow[d] = (h16)(nv * (1.0f + adb[scOff + d]) + adb[shOff + d]);
    }
}

// ---------------- tiled WMMA GEMM:  out[M,N] = A[M,K] @ W[N,K]^T  ----------------
// 256 threads = 8 waves; block tile 128x128; wave tile 32x64; K staged 64 at a time,
// double-buffered through registers so global loads overlap WMMA.
// EPI: 0 = store f16                  (QKV)
//      1 = f32: resid + gate*acc      (W_out)
//      2 = f16: gelu(acc + bias)      (MLP1)
//      3 = f32: resid + gate*(acc+b)  (MLP2 -> d_out)
template <int EPI>
__global__ __launch_bounds__(256) void k_gemm(const h16* __restrict__ A,
                                              const h16* __restrict__ W,
                                              const float* __restrict__ bias,
                                              const float* __restrict__ resid,
                                              const float* __restrict__ gate,
                                              float* __restrict__ outF,
                                              h16* __restrict__ outH,
                                              int N, int K) {
    __shared__ h16 As[128 * 72];   // 64-K slice, padded to 72 halfs/row
    __shared__ h16 Ws[128 * 72];

    const int tid = threadIdx.x;
    const int lane = tid & 31;
    const int wave = tid >> 5;
    const int lane16 = lane & 15;
    const int g = lane >> 4;
    const int waveM = wave >> 1, waveN = wave & 1;
    const int bM = blockIdx.y * 128, bN = blockIdx.x * 128;

    v8f acc[2][4];
#pragma unroll
    for (int mi = 0; mi < 2; mi++)
#pragma unroll
        for (int ni = 0; ni < 4; ni++)
#pragma unroll
            for (int r = 0; r < 8; r++) acc[mi][ni][r] = 0.0f;

    const int ldRow  = tid >> 1;          // 0..127
    const int ldPart = (tid & 1) * 32;    // half offset 0 or 32

    // preload first K slice into registers
    f4 ra[4], rw[4];
    {
        const f4* ga = reinterpret_cast<const f4*>(A + (size_t)(bM + ldRow) * K + ldPart);
        const f4* gw = reinterpret_cast<const f4*>(W + (size_t)(bN + ldRow) * K + ldPart);
#pragma unroll
        for (int j = 0; j < 4; j++) { ra[j] = ga[j]; rw[j] = gw[j]; }
    }

    for (int kk = 0; kk < K; kk += 64) {
        f4* la = reinterpret_cast<f4*>(&As[ldRow * 72 + ldPart]);
        f4* lw = reinterpret_cast<f4*>(&Ws[ldRow * 72 + ldPart]);
#pragma unroll
        for (int j = 0; j < 4; j++) { la[j] = ra[j]; lw[j] = rw[j]; }
        __syncthreads();

        if (kk + 64 < K) {   // prefetch next slice while WMMAs run
            const f4* ga = reinterpret_cast<const f4*>(A + (size_t)(bM + ldRow) * K + kk + 64 + ldPart);
            const f4* gw = reinterpret_cast<const f4*>(W + (size_t)(bN + ldRow) * K + kk + 64 + ldPart);
#pragma unroll
            for (int j = 0; j < 4; j++) { ra[j] = ga[j]; rw[j] = gw[j]; }
        }

#pragma unroll
        for (int kt = 0; kt < 2; kt++) {
            Frag af[2], bf[4];
#pragma unroll
            for (int mi = 0; mi < 2; mi++) {
                int r = waveM * 32 + mi * 16 + lane16;
                af[mi].f[0] = *reinterpret_cast<const f4*>(&As[r * 72 + kt * 32 + 8 * g]);
                af[mi].f[1] = *reinterpret_cast<const f4*>(&As[r * 72 + kt * 32 + 16 + 8 * g]);
            }
#pragma unroll
            for (int ni = 0; ni < 4; ni++) {
                int r = waveN * 64 + ni * 16 + lane16;
                bf[ni].f[0] = *reinterpret_cast<const f4*>(&Ws[r * 72 + kt * 32 + 8 * g]);
                bf[ni].f[1] = *reinterpret_cast<const f4*>(&Ws[r * 72 + kt * 32 + 16 + 8 * g]);
            }
#pragma unroll
            for (int mi = 0; mi < 2; mi++)
#pragma unroll
                for (int ni = 0; ni < 4; ni++)
                    acc[mi][ni] = wmma_f16(af[mi].h, bf[ni].h, acc[mi][ni]);
        }
        __syncthreads();
    }

    // ---- epilogue ----
#pragma unroll
    for (int mi = 0; mi < 2; mi++) {
#pragma unroll
        for (int ni = 0; ni < 4; ni++) {
            int colG = bN + waveN * 64 + ni * 16 + lane16;
#pragma unroll
            for (int r = 0; r < 8; r++) {
                int rowG = bM + waveM * 32 + mi * 16 + r + 8 * g;
                float v = acc[mi][ni][r];
                if (EPI == 2 || EPI == 3) v += bias[colG];
                if (EPI == 0) {
                    outH[(size_t)rowG * N + colG] = (h16)v;
                } else if (EPI == 2) {
                    float x3 = v * v * v;
                    float gl = 0.5f * v * (1.0f + tanhf(0.7978845608f * (v + 0.044715f * x3)));
                    outH[(size_t)rowG * N + colG] = (h16)gl;
                } else {
                    int b = rowG >> 11;
                    outF[(size_t)rowG * N + colG] =
                        resid[(size_t)rowG * N + colG] + gate[b * SIX_ + colG] * v;
                }
            }
        }
    }
}

// ---------------- RoPE + layout shuffle ----------------
// qkv (B,S,3,H,DH) f16 -> q (scaled by 1/sqrt(DH)), k as (B,H,S,DH), v^T (B,H,DH,S)
__global__ void k_rope(const h16* __restrict__ qkv, const float* __restrict__ cosT,
                       const float* __restrict__ sinT, h16* __restrict__ q,
                       h16* __restrict__ k, h16* __restrict__ vt) {
    int i = blockIdx.x * blockDim.x + threadIdx.x;
    if (i >= B_ * S_ * H_ * 32) return;
    int d2 = i & 31;
    int h  = (i >> 5) % H_;
    int s  = (i / (32 * H_)) % S_;
    int b  = i / (32 * H_ * S_);
    size_t base = (size_t)(b * S_ + s) * (3 * D_);
    // cos input shape (1,S,3,1,DH): take rep=0, d<32
    float cv = cosT[s * 3 * DH_ + d2];
    float sv = sinT[s * 3 * DH_ + d2];
    const float qs = 0.125f;   // 1/sqrt(64) folded into q

    size_t qb = (size_t)((b * H_ + h) * S_ + s) * DH_;
    float q1 = (float)qkv[base + h * DH_ + d2];
    float q2 = (float)qkv[base + h * DH_ + d2 + 32];
    q[qb + d2]      = (h16)((q1 * cv - q2 * sv) * qs);
    q[qb + d2 + 32] = (h16)((q1 * sv + q2 * cv) * qs);

    float k1 = (float)qkv[base + D_ + h * DH_ + d2];
    float k2 = (float)qkv[base + D_ + h * DH_ + d2 + 32];
    k[qb + d2]      = (h16)(k1 * cv - k2 * sv);
    k[qb + d2 + 32] = (h16)(k1 * sv + k2 * cv);

    size_t vb = (size_t)((b * H_ + h) * DH_) * S_ + s;
    vt[vb + (size_t)d2 * S_]        = qkv[base + 2 * D_ + h * DH_ + d2];
    vt[vb + (size_t)(d2 + 32) * S_] = qkv[base + 2 * D_ + h * DH_ + d2 + 32];
}

// ---------------- flash attention ----------------
// grid (S/128, H, B), 128 threads = 4 waves, each wave owns 32 q rows.
__global__ __launch_bounds__(128) void k_attn(const h16* __restrict__ Q,
                                              const h16* __restrict__ Kt,
                                              const h16* __restrict__ Vt,
                                              h16* __restrict__ O) {
    __shared__ h16 Ks[64 * 72];        // keys x dh
    __shared__ h16 Vs[64 * 72];        // dh x keys  (V^T)
    __shared__ h16 Ps[4 * 32 * 72];    // per-wave P scratch

    const int tid = threadIdx.x;
    const int lane = tid & 31;
    const int wave = tid >> 5;
    const int lane16 = lane & 15;
    const int g = lane >> 4;
    const int qb = blockIdx.x;
    const int h  = blockIdx.y;
    const int b  = blockIdx.z;
    const int bh = b * H_ + h;

    // Q fragments held in registers for whole K loop (already pre-scaled by 1/8)
    Frag qf[2][2];
#pragma unroll
    for (int mi = 0; mi < 2; mi++) {
        int qrow = qb * 128 + wave * 32 + mi * 16 + lane16;
        const h16* qp = Q + ((size_t)bh * S_ + qrow) * DH_;
#pragma unroll
        for (int kt = 0; kt < 2; kt++) {
            qf[mi][kt].f[0] = *reinterpret_cast<const f4*>(qp + kt * 32 + 8 * g);
            qf[mi][kt].f[1] = *reinterpret_cast<const f4*>(qp + kt * 32 + 16 + 8 * g);
        }
    }

    v8f oacc[2][4];
    float mstat[2][8], lstat[2][8];
#pragma unroll
    for (int mi = 0; mi < 2; mi++) {
#pragma unroll
        for (int ni = 0; ni < 4; ni++)
#pragma unroll
            for (int r = 0; r < 8; r++) oacc[mi][ni][r] = 0.0f;
#pragma unroll
        for (int r = 0; r < 8; r++) { mstat[mi][r] = -3.0e38f; lstat[mi][r] = 0.0f; }
    }

    const int ldRow  = tid & 63;
    const int ldPart = (tid >> 6) * 32;
    h16* pw = Ps + wave * (32 * 72);

    // preload first K/V tile into registers
    f4 rk[4], rv[4];
    {
        const f4* kp = reinterpret_cast<const f4*>(
            Kt + ((size_t)bh * S_ + ldRow) * DH_ + ldPart);
        const f4* vp = reinterpret_cast<const f4*>(
            Vt + ((size_t)bh * DH_ + ldRow) * S_ + ldPart);
#pragma unroll
        for (int j = 0; j < 4; j++) { rk[j] = kp[j]; rv[j] = vp[j]; }
    }

    for (int kb = 0; kb < S_ / 64; kb++) {
        {
            f4* dk = reinterpret_cast<f4*>(&Ks[ldRow * 72 + ldPart]);
            f4* dv = reinterpret_cast<f4*>(&Vs[ldRow * 72 + ldPart]);
#pragma unroll
            for (int j = 0; j < 4; j++) { dk[j] = rk[j]; dv[j] = rv[j]; }
        }
        __syncthreads();

        if (kb + 1 < S_ / 64) {   // prefetch next K/V tile during compute
            const f4* kp = reinterpret_cast<const f4*>(
                Kt + ((size_t)bh * S_ + (size_t)(kb + 1) * 64 + ldRow) * DH_ + ldPart);
            const f4* vp = reinterpret_cast<const f4*>(
                Vt + ((size_t)bh * DH_ + ldRow) * S_ + (size_t)(kb + 1) * 64 + ldPart);
#pragma unroll
            for (int j = 0; j < 4; j++) { rk[j] = kp[j]; rv[j] = vp[j]; }
        }

        // S = Q K^T, online softmax (DPP reductions), P -> LDS
#pragma unroll
        for (int mi = 0; mi < 2; mi++) {
            v8f sc_[4];
#pragma unroll
            for (int ni = 0; ni < 4; ni++)
#pragma unroll
                for (int r = 0; r < 8; r++) sc_[ni][r] = 0.0f;
            // load 4 K fragments of a kt-slice into distinct regs, then burst WMMAs
#pragma unroll
            for (int kt = 0; kt < 2; kt++) {
                Frag kf4[4];
#pragma unroll
                for (int ni = 0; ni < 4; ni++) {
                    int rr = ni * 16 + lane16;
                    kf4[ni].f[0] = *reinterpret_cast<const f4*>(&Ks[rr * 72 + kt * 32 + 8 * g]);
                    kf4[ni].f[1] = *reinterpret_cast<const f4*>(&Ks[rr * 72 + kt * 32 + 16 + 8 * g]);
                }
#pragma unroll
                for (int ni = 0; ni < 4; ni++)
                    sc_[ni] = wmma_f16(qf[mi][kt].h, kf4[ni].h, sc_[ni]);
            }
#pragma unroll
            for (int r = 0; r < 8; r++) {
                float rmax = fmaxf(fmaxf(sc_[0][r], sc_[1][r]), fmaxf(sc_[2][r], sc_[3][r]));
                rmax = red16_max(rmax);
                float mo  = mstat[mi][r];
                float mn  = fmaxf(mo, rmax);
                float esc = __expf(mo - mn);
                float rs  = 0.0f;
#pragma unroll
                for (int ni = 0; ni < 4; ni++) {
                    float p = __expf(sc_[ni][r] - mn);
                    sc_[ni][r] = p;
                    rs += p;
                }
                rs = red16_sum(rs);
                mstat[mi][r] = mn;
                lstat[mi][r] = lstat[mi][r] * esc + rs;
#pragma unroll
                for (int ni = 0; ni < 4; ni++) oacc[mi][ni][r] *= esc;
                int prow = mi * 16 + r + 8 * g;
#pragma unroll
                for (int ni = 0; ni < 4; ni++)
                    pw[prow * 72 + ni * 16 + lane16] = (h16)sc_[ni][r];
            }
        }

        // O += P V   (P and V fragments preloaded per kt slice)
#pragma unroll
        for (int kt = 0; kt < 2; kt++) {
            Frag pf[2], vf[4];
#pragma unroll
            for (int mi = 0; mi < 2; mi++) {
                int rr = mi * 16 + lane16;
                pf[mi].f[0] = *reinterpret_cast<const f4*>(&pw[rr * 72 + kt * 32 + 8 * g]);
                pf[mi].f[1] = *reinterpret_cast<const f4*>(&pw[rr * 72 + kt * 32 + 16 + 8 * g]);
            }
#pragma unroll
            for (int ni = 0; ni < 4; ni++) {
                int vr = ni * 16 + lane16;
                vf[ni].f[0] = *reinterpret_cast<const f4*>(&Vs[vr * 72 + kt * 32 + 8 * g]);
                vf[ni].f[1] = *reinterpret_cast<const f4*>(&Vs[vr * 72 + kt * 32 + 16 + 8 * g]);
            }
#pragma unroll
            for (int mi = 0; mi < 2; mi++)
#pragma unroll
                for (int ni = 0; ni < 4; ni++)
                    oacc[mi][ni] = wmma_f16(pf[mi].h, vf[ni].h, oacc[mi][ni]);
        }
        __syncthreads();
    }

    // write O as (b, s, h*64+dh) f16, row-major (8192, 768)
    float linv[2][8];
#pragma unroll
    for (int mi = 0; mi < 2; mi++)
#pragma unroll
        for (int r = 0; r < 8; r++) linv[mi][r] = __frcp_rn(lstat[mi][r]);
#pragma unroll
    for (int mi = 0; mi < 2; mi++) {
#pragma unroll
        for (int ni = 0; ni < 4; ni++) {
            int dh = ni * 16 + lane16;
#pragma unroll
            for (int r = 0; r < 8; r++) {
                int qrow = qb * 128 + wave * 32 + mi * 16 + r + 8 * g;
                float ov = oacc[mi][ni][r] * linv[mi][r];
                O[(size_t)(b * S_ + qrow) * D_ + h * DH_ + dh] = (h16)ov;
            }
        }
    }
}

// ---------------- orchestration ----------------
extern "C" void kernel_launch(void* const* d_in, const int* in_sizes, int n_in,
                              void* d_out, int out_size, void* d_ws, size_t ws_size,
                              hipStream_t stream) {
    (void)in_sizes; (void)n_in; (void)out_size; (void)ws_size;
    const float* x      = (const float*)d_in[0];
    const float* cosT   = (const float*)d_in[1];
    const float* sinT   = (const float*)d_in[2];
    const float* c      = (const float*)d_in[3];
    const float* ln1_w  = (const float*)d_in[4];
    const float* W_qkv  = (const float*)d_in[5];
    const float* W_out  = (const float*)d_in[6];
    const float* W_mlp1 = (const float*)d_in[7];
    const float* b_mlp1 = (const float*)d_in[8];
    const float* W_mlp2 = (const float*)d_in[9];
    const float* b_mlp2 = (const float*)d_in[10];
    const float* ln2_w  = (const float*)d_in[11];
    const float* W_ada  = (const float*)d_in[12];
    const float* b_ada  = (const float*)d_in[13];
    float* out = (float*)d_out;

    char* ws = (char*)d_ws;
    size_t off = 0;
    auto alloc = [&](size_t bytes) {
        char* p = ws + off;
        off += (bytes + 255) & ~(size_t)255;
        return p;
    };
    h16*   w16_qkv = (h16*)alloc((size_t)3 * D_ * D_ * 2);
    h16*   w16_out = (h16*)alloc((size_t)D_ * D_ * 2);
    h16*   w16_m1  = (h16*)alloc((size_t)MLP_ * D_ * 2);
    h16*   w16_m2  = (h16*)alloc((size_t)D_ * MLP_ * 2);
    float* ada     = (float*)alloc((size_t)B_ * SIX_ * 4);
    h16*   hbuf    = (h16*)alloc((size_t)MROWS * D_ * 2);     // h, then h2
    h16*   big     = (h16*)alloc((size_t)MROWS * MLP_ * 2);   // qkv, then m1
    h16*   q16     = (h16*)alloc((size_t)B_ * H_ * S_ * DH_ * 2);
    h16*   k16     = (h16*)alloc((size_t)B_ * H_ * S_ * DH_ * 2);
    h16*   vt16    = (h16*)alloc((size_t)B_ * H_ * DH_ * S_ * 2);
    h16*   o16     = (h16*)alloc((size_t)MROWS * D_ * 2);
    float* x1      = (float*)alloc((size_t)MROWS * D_ * 4);

    int n;
    n = 3 * D_ * D_; k_f32_to_f16<<<(n + 255) / 256, 256, 0, stream>>>(W_qkv, w16_qkv, n);
    n = D_ * D_;     k_f32_to_f16<<<(n + 255) / 256, 256, 0, stream>>>(W_out, w16_out, n);
    n = MLP_ * D_;   k_f32_to_f16<<<(n + 255) / 256, 256, 0, stream>>>(W_mlp1, w16_m1, n);
    n = D_ * MLP_;   k_f32_to_f16<<<(n + 255) / 256, 256, 0, stream>>>(W_mlp2, w16_m2, n);

    k_ada<<<(B_ * SIX_ + 255) / 256, 256, 0, stream>>>(c, W_ada, b_ada, ada);

    // h = ln1(x)*(1+sc_msa)+sh_msa   (sh at 0, sc at D)
    k_ln_mod<<<MROWS, 256, 0, stream>>>(x, ln1_w, ada, 0, D_, hbuf);

    // qkv = h @ W_qkv^T  -> f16
    k_gemm<0><<<dim3(3 * D_ / 128, MROWS / 128), 256, 0, stream>>>(
        hbuf, w16_qkv, nullptr, nullptr, nullptr, nullptr, big, 3 * D_, D_);

    k_rope<<<(B_ * S_ * H_ * 32 + 255) / 256, 256, 0, stream>>>(big, cosT, sinT, q16, k16, vt16);

    k_attn<<<dim3(S_ / 128, H_, B_), 128, 0, stream>>>(q16, k16, vt16, o16);

    // x1 = x + g_msa * (o @ W_out^T)   (gate = ada + 2D)
    k_gemm<1><<<dim3(D_ / 128, MROWS / 128), 256, 0, stream>>>(
        o16, w16_out, nullptr, x, ada + 2 * D_, x1, nullptr, D_, D_);

    // h2 = ln2(x1)*(1+sc_mlp)+sh_mlp  (sh at 3D, sc at 4D)
    k_ln_mod<<<MROWS, 256, 0, stream>>>(x1, ln2_w, ada, 3 * D_, 4 * D_, hbuf);

    // m1 = gelu(h2 @ W_mlp1^T + b1) -> f16
    k_gemm<2><<<dim3(MLP_ / 128, MROWS / 128), 256, 0, stream>>>(
        hbuf, w16_m1, b_mlp1, nullptr, nullptr, nullptr, big, MLP_, D_);

    // out = x1 + g_mlp * (m1 @ W_mlp2^T + b2)   (gate = ada + 5D)
    k_gemm<3><<<dim3(D_ / 128, MROWS / 128), 256, 0, stream>>>(
        big, w16_m2, b_mlp2, x1, ada + 5 * D_, out, nullptr, D_, MLP_);
}
